// DecoderLayer_82523501625535
// MI455X (gfx1250) — matrix operand order repaired and checked by reading, not compiled, not run on hardware
//
#include <hip/hip_runtime.h>

// ---- problem constants ----
#define S_LEN    2048
#define DMODEL   1024
#define N_HEADS  16
#define D_HEAD   64
#define D_HID    4096
#define BATCH    2
#define ROWS     (BATCH * S_LEN)       // 4096
#define MASK_VAL 1.0e10f
#define INV_SCALE 0.03125f             // 1/sqrt(1024)

typedef __attribute__((ext_vector_type(16))) __bf16 v16bf;
typedef __attribute__((ext_vector_type(8)))  __bf16 v8bf;
typedef __attribute__((ext_vector_type(8)))  float  v8f;

union BF16Frag { v16bf v; v8bf h[2]; };

__device__ __forceinline__ v8f wmma_bf16(v16bf a, v16bf b, v8f c) {
  return __builtin_amdgcn_wmma_f32_16x16x32_bf16(false, a, false, b, (short)0, c, false, false);
}

// ---------------- fp32 -> bf16 convert ----------------
__global__ __launch_bounds__(256) void k_cvt(const float* __restrict__ in,
                                             __bf16* __restrict__ out, int n) {
  int i = blockIdx.x * 256 + threadIdx.x;
  if (i < n) out[i] = (__bf16)in[i];
}

// ---------------- V transpose: [B*S, D] -> [B,H,D_HEAD,S] ----------------
__global__ __launch_bounds__(256) void k_transpose_v(const __bf16* __restrict__ v,
                                                     __bf16* __restrict__ vt) {
  int idx = blockIdx.x * 256 + threadIdx.x;          // over B*H*64*2048 = 4M
  if (idx >= BATCH * N_HEADS * D_HEAD * S_LEN) return;
  int s = idx & (S_LEN - 1);
  int d = (idx >> 11) & (D_HEAD - 1);
  int h = (idx >> 17) & (N_HEADS - 1);
  int b = idx >> 21;
  vt[idx] = v[((size_t)(b * S_LEN + s)) * DMODEL + h * D_HEAD + d];
}

// ---------------- WMMA GEMM: C[M,N] = A[M,K] @ W[N,K]^T (+bias)(+relu) ----------------
// block = 256 threads = 8 waves; wave computes 32x64 (8 WMMAs per 32-K step, B reused 2x);
// block computes 256x64. K-loop unrolled by 64 with ping-pong fragment sets (no copies).
// Requires K % 64 == 0 (holds: K is 1024 or 4096).
template <typename OutT, bool RELU>
__global__ __launch_bounds__(256)
void k_gemm(const __bf16* __restrict__ A, const __bf16* __restrict__ W,
            const float* __restrict__ bias, OutT* __restrict__ C,
            int M, int N, int K) {
  const int lane = threadIdx.x & 31;
  const int wave = threadIdx.x >> 5;
  const int l15  = lane & 15;
  const int hi   = lane >> 4;               // 0 or 1
  const int mBase = blockIdx.y * 256 + wave * 32;
  const int nBase = blockIdx.x * 64;

  v8f acc[2][4] = {};
  const __bf16* aRow0 = A + (size_t)(mBase + l15) * K;       // rows mBase..+15
  const __bf16* aRow1 = A + (size_t)(mBase + 16 + l15) * K;  // rows mBase+16..+31
  const int aKoff = hi * 8;  // A frag: lane<16 -> K{0-7,16-23}; lane>=16 -> K{8-15,24-31}

  auto loadA = [&](BF16Frag& f, const __bf16* r, int k0) {
    f.h[0] = *reinterpret_cast<const v8bf*>(r + k0 + aKoff);
    f.h[1] = *reinterpret_cast<const v8bf*>(r + k0 + 16 + aKoff);
  };
  auto loadB = [&](BF16Frag& f, int t, int k0) {
    // B frag (32x16): lane holds column n = row of W, K{8h..8h+7, 16+8h..16+8h+7}
    const __bf16* wr = W + (size_t)(nBase + t * 16 + l15) * K + k0 + hi * 8;
    f.h[0] = *reinterpret_cast<const v8bf*>(wr);
    f.h[1] = *reinterpret_cast<const v8bf*>(wr + 16);
  };

  BF16Frag a0A, a1A, bA[4];                 // ping
  BF16Frag a0B, a1B, bB[4];                 // pong
  loadA(a0A, aRow0, 0);
  loadA(a1A, aRow1, 0);
#pragma unroll
  for (int t = 0; t < 4; ++t) loadB(bA[t], t, 0);

  int k = 0;
  while (true) {
    const int k1 = k + 32;
    __builtin_prefetch(aRow0 + k1 + 96, 0, 1);
    loadA(a0B, aRow0, k1);                  // load pong (in flight under ping WMMAs)
    loadA(a1B, aRow1, k1);
#pragma unroll
    for (int t = 0; t < 4; ++t) loadB(bB[t], t, k1);

#pragma unroll
    for (int t = 0; t < 4; ++t) {           // compute ping
      acc[0][t] = wmma_bf16(a0A.v, bA[t].v, acc[0][t]);
      acc[1][t] = wmma_bf16(a1A.v, bA[t].v, acc[1][t]);
    }

    const int k2 = k + 64;
    if (k2 < K) {                           // reload ping (under pong WMMAs)
      loadA(a0A, aRow0, k2);
      loadA(a1A, aRow1, k2);
#pragma unroll
      for (int t = 0; t < 4; ++t) loadB(bA[t], t, k2);
    }

#pragma unroll
    for (int t = 0; t < 4; ++t) {           // compute pong
      acc[0][t] = wmma_bf16(a0B.v, bB[t].v, acc[0][t]);
      acc[1][t] = wmma_bf16(a1B.v, bB[t].v, acc[1][t]);
    }

    if (k2 >= K) break;
    k = k2;
  }

#pragma unroll
  for (int half = 0; half < 2; ++half) {
    const int rowBase = mBase + half * 16 + hi * 8;
#pragma unroll
    for (int t = 0; t < 4; ++t) {
      const int col = nBase + t * 16 + l15;
      const float bv = bias ? bias[col] : 0.0f;
#pragma unroll
      for (int v = 0; v < 8; ++v) {
        float val = acc[half][t][v] + bv;
        if (RELU) val = fmaxf(val, 0.0f);
        C[(size_t)(rowBase + v) * N + col] = (OutT)val;
      }
    }
  }
}

// ---------------- flash attention: one wave per (b, h, 16-query tile) ----------------
// Q,K: [B*S, D] bf16 (head h at cols h*64..); VT: [B,H,64,S] bf16; O: [B*S, D] bf16
template <bool CAUSAL>
__global__ __launch_bounds__(256)
void k_attn(const __bf16* __restrict__ Q, const __bf16* __restrict__ Km,
            const __bf16* __restrict__ VT, __bf16* __restrict__ O) {
  __shared__ __bf16 ldsP[8][16 * 32];       // per-wave P tile staging (16x32)
  const int lane = threadIdx.x & 31;
  const int wave = threadIdx.x >> 5;
  const int l15  = lane & 15;
  const int hi   = lane >> 4;

  const int QT = S_LEN / 16;
  const int gw = blockIdx.x * 8 + wave;
  const int qt = gw % QT;
  const int h  = (gw / QT) % N_HEADS;
  const int b  = gw / (QT * N_HEADS);
  const int qbase = qt * 16;

  // Q fragments (A layout, 16x32 over d_h chunks 0..31 and 32..63)
  BF16Frag qa0, qa1;
  {
    const __bf16* qrow = Q + ((size_t)(b * S_LEN + qbase + l15)) * DMODEL + h * D_HEAD;
    const int koff = hi * 8;
    qa0.h[0] = *reinterpret_cast<const v8bf*>(qrow + koff);
    qa0.h[1] = *reinterpret_cast<const v8bf*>(qrow + 16 + koff);
    qa1.h[0] = *reinterpret_cast<const v8bf*>(qrow + 32 + koff);
    qa1.h[1] = *reinterpret_cast<const v8bf*>(qrow + 48 + koff);
  }

  float rm[8], rl[8];
#pragma unroll
  for (int v = 0; v < 8; ++v) { rm[v] = -3.0e38f; rl[v] = 0.0f; }
  v8f acc[4] = {};

  __bf16* myP = ldsP[wave];
  const __bf16* vtb = VT + ((size_t)(b * N_HEADS + h)) * D_HEAD * S_LEN;
  const int kmax = CAUSAL ? (qbase + 16) : S_LEN;

  for (int k0 = 0; k0 < kmax; k0 += 32) {
    v8f s[2];
#pragma unroll
    for (int j = 0; j < 2; ++j) {
      const int key = k0 + j * 16 + l15;
      const __bf16* krow = Km + ((size_t)(b * S_LEN + key)) * DMODEL + h * D_HEAD;
      BF16Frag kb0, kb1;                    // B frags (32x16): d_h {8h..+7, 16+8h..+7}
      kb0.h[0] = *reinterpret_cast<const v8bf*>(krow + hi * 8);
      kb0.h[1] = *reinterpret_cast<const v8bf*>(krow + 16 + hi * 8);
      kb1.h[0] = *reinterpret_cast<const v8bf*>(krow + 32 + hi * 8);
      kb1.h[1] = *reinterpret_cast<const v8bf*>(krow + 48 + hi * 8);
      v8f sv = {};
      sv = wmma_bf16(qa0.v, kb0.v, sv);
      sv = wmma_bf16(qa1.v, kb1.v, sv);
#pragma unroll
      for (int v = 0; v < 8; ++v) {
        float x = sv[v];
        if (CAUSAL) {
          const int qrow_g = qbase + v + hi * 8;
          if (key > qrow_g) x -= MASK_VAL;  // mask BEFORE scaling, as reference
        }
        sv[v] = x * INV_SCALE;
      }
      s[j] = sv;
    }

    // online softmax update (row = v + hi*8, replicated across a 16-lane group)
#pragma unroll
    for (int v = 0; v < 8; ++v) {
      float bm = fmaxf(s[0][v], s[1][v]);
      bm = fmaxf(bm, __shfl_xor(bm, 1, 32));
      bm = fmaxf(bm, __shfl_xor(bm, 2, 32));
      bm = fmaxf(bm, __shfl_xor(bm, 4, 32));
      bm = fmaxf(bm, __shfl_xor(bm, 8, 32));
      const float mnew = fmaxf(rm[v], bm);
      const float sc = __expf(rm[v] - mnew);
      rm[v] = mnew;
      const float p0 = __expf(s[0][v] - mnew);
      const float p1 = __expf(s[1][v] - mnew);
      float ps = p0 + p1;
      ps += __shfl_xor(ps, 1, 32);
      ps += __shfl_xor(ps, 2, 32);
      ps += __shfl_xor(ps, 4, 32);
      ps += __shfl_xor(ps, 8, 32);
      rl[v] = rl[v] * sc + ps;
#pragma unroll
      for (int t = 0; t < 4; ++t) acc[t][v] *= sc;
      const int m = v + hi * 8;
      myP[m * 32 + l15]      = (__bf16)p0;  // stage P (16x32) in row-major LDS
      myP[m * 32 + 16 + l15] = (__bf16)p1;
    }

    // reload P as A fragment (same-wave LDS ops are in-order)
    BF16Frag pa;
    {
      const int koff = hi * 8;
      pa.h[0] = *reinterpret_cast<const v8bf*>(myP + l15 * 32 + koff);
      pa.h[1] = *reinterpret_cast<const v8bf*>(myP + l15 * 32 + 16 + koff);
    }
    // P(16x32) @ V(32x64) via 4 WMMAs; VT gives contiguous B-frag loads
#pragma unroll
    for (int t = 0; t < 4; ++t) {
      BF16Frag vb;
      vb.h[0] = *reinterpret_cast<const v8bf*>(vtb + (size_t)(t * 16 + l15) * S_LEN + k0 + hi * 8);
      vb.h[1] = *reinterpret_cast<const v8bf*>(vtb + (size_t)(t * 16 + l15) * S_LEN + k0 + 16 + hi * 8);
      acc[t] = wmma_bf16(pa.v, vb.v, acc[t]);
    }
  }

#pragma unroll
  for (int t = 0; t < 4; ++t)
#pragma unroll
    for (int v = 0; v < 8; ++v) {
      const int m = v + hi * 8;
      const float o = acc[t][v] / rl[v];
      O[((size_t)(b * S_LEN + qbase + m)) * DMODEL + h * D_HEAD + t * 16 + l15] = (__bf16)o;
    }
}

// ---------------- residual + LayerNorm (unbiased std, /(std+eps)) ----------------
__global__ __launch_bounds__(256)
void k_add_ln(const float* __restrict__ x, const float* __restrict__ y,
              const float* __restrict__ g, const float* __restrict__ bt,
              float* __restrict__ outf, __bf16* __restrict__ outbf) {
  __shared__ float red[8];
  const int row = blockIdx.x;
  const int lane = threadIdx.x & 31, wave = threadIdx.x >> 5;
  const float* xr = x + (size_t)row * DMODEL;
  const float* yr = y + (size_t)row * DMODEL;

  float vals[4];
  float s = 0.0f;
#pragma unroll
  for (int i = 0; i < 4; ++i) {
    const int c = threadIdx.x + i * 256;
    vals[i] = xr[c] + yr[c];
    s += vals[i];
  }
  for (int m = 1; m < 32; m <<= 1) s += __shfl_xor(s, m, 32);
  if (lane == 0) red[wave] = s;
  __syncthreads();
  float tot = 0.0f;
#pragma unroll
  for (int w = 0; w < 8; ++w) tot += red[w];
  const float mean = tot / (float)DMODEL;
  __syncthreads();

  float sq = 0.0f;
#pragma unroll
  for (int i = 0; i < 4; ++i) { const float d = vals[i] - mean; sq += d * d; }
  for (int m = 1; m < 32; m <<= 1) sq += __shfl_xor(sq, m, 32);
  if (lane == 0) red[wave] = sq;
  __syncthreads();
  float tsq = 0.0f;
#pragma unroll
  for (int w = 0; w < 8; ++w) tsq += red[w];
  const float stddev = sqrtf(tsq / (float)(DMODEL - 1));
  const float inv = 1.0f / (stddev + 1e-6f);

#pragma unroll
  for (int i = 0; i < 4; ++i) {
    const int c = threadIdx.x + i * 256;
    const float o = g[c] * (vals[i] - mean) * inv + bt[c];
    outf[(size_t)row * DMODEL + c] = o;
    if (outbf) outbf[(size_t)row * DMODEL + c] = (__bf16)o;
  }
}

// ---------------- host-side orchestration ----------------
extern "C" void kernel_launch(void* const* d_in, const int* in_sizes, int n_in,
                              void* d_out, int out_size, void* d_ws, size_t ws_size,
                              hipStream_t stream) {
  const float* x    = (const float*)d_in[0];
  const float* enc  = (const float*)d_in[1];
  const float* wsaf[4] = {(const float*)d_in[2], (const float*)d_in[3],
                          (const float*)d_in[4], (const float*)d_in[5]};
  const float* wcaf[4] = {(const float*)d_in[6], (const float*)d_in[7],
                          (const float*)d_in[8], (const float*)d_in[9]};
  const float* ffw1 = (const float*)d_in[10];
  const float* ffb1 = (const float*)d_in[11];
  const float* ffw2 = (const float*)d_in[12];
  const float* ffb2 = (const float*)d_in[13];
  const float* ln1g = (const float*)d_in[14];
  const float* ln1b = (const float*)d_in[15];
  const float* ln2g = (const float*)d_in[16];
  const float* ln2b = (const float*)d_in[17];
  const float* ln3g = (const float*)d_in[18];
  const float* ln3b = (const float*)d_in[19];

  const size_t E1 = (size_t)DMODEL * DMODEL;     // 1M elems (DxD weight)
  const size_t EA = (size_t)ROWS * DMODEL;       // 4M elems (activation)
  const size_t EW = (size_t)D_HID * DMODEL;      // 4M elems (FFN weight)

  char* ws = (char*)d_ws;
  size_t off = 0;
  auto take = [&](size_t bytes) { void* p = ws + off; off += bytes; return p; };

  __bf16* wsab[4]; for (int i = 0; i < 4; ++i) wsab[i] = (__bf16*)take(E1 * 2);
  __bf16* wcab[4]; for (int i = 0; i < 4; ++i) wcab[i] = (__bf16*)take(E1 * 2);
  __bf16* w1b  = (__bf16*)take(EW * 2);
  __bf16* w2b  = (__bf16*)take(EW * 2);
  __bf16* encb = (__bf16*)take(EA * 2);
  __bf16* actb = (__bf16*)take(EA * 2);          // x -> h1 -> h2 (bf16 operand)
  __bf16* qb   = (__bf16*)take(EA * 2);          // contiguous: qb,kb,vb,vtb
  __bf16* kb   = (__bf16*)take(EA * 2);
  __bf16* vb   = (__bf16*)take(EA * 2);
  __bf16* vtb  = (__bf16*)take(EA * 2);
  __bf16* ff1  = qb;                             // stage-C alias: 16M elems across qb..vtb
  __bf16* attnb = (__bf16*)take(EA * 2);
  float*  tmpf = (float*)take(EA * 4);
  float*  hf   = (float*)take(EA * 4);
  (void)ws_size; (void)in_sizes; (void)n_in; (void)out_size;

  auto cvt = [&](const float* s, __bf16* d, size_t n) {
    k_cvt<<<(unsigned)((n + 255) / 256), 256, 0, stream>>>(s, d, (int)n);
  };

  // convert inputs + all weights to bf16 (recomputed every call: deterministic)
  cvt(x, actb, EA);
  cvt(enc, encb, EA);
  for (int i = 0; i < 4; ++i) cvt(wsaf[i], wsab[i], E1);
  for (int i = 0; i < 4; ++i) cvt(wcaf[i], wcab[i], E1);
  cvt(ffw1, w1b, EW);
  cvt(ffw2, w2b, EW);

  const dim3 gDD(DMODEL / 64, ROWS / 256);       // (16,16)
  const dim3 gDH(D_HID / 64, ROWS / 256);        // (64,16)

  // ---- stage A: causal self-attention ----
  k_gemm<__bf16, false><<<gDD, 256, 0, stream>>>(actb, wsab[0], nullptr, qb, ROWS, DMODEL, DMODEL);
  k_gemm<__bf16, false><<<gDD, 256, 0, stream>>>(actb, wsab[1], nullptr, kb, ROWS, DMODEL, DMODEL);
  k_gemm<__bf16, false><<<gDD, 256, 0, stream>>>(actb, wsab[2], nullptr, vb, ROWS, DMODEL, DMODEL);
  k_transpose_v<<<(unsigned)(EA / 256), 256, 0, stream>>>(vb, vtb);
  k_attn<true><<<BATCH * N_HEADS * (S_LEN / 16) / 8, 256, 0, stream>>>(qb, kb, vtb, attnb);
  k_gemm<float, false><<<gDD, 256, 0, stream>>>(attnb, wsab[3], nullptr, tmpf, ROWS, DMODEL, DMODEL);
  k_add_ln<<<ROWS, 256, 0, stream>>>(x, tmpf, ln1g, ln1b, hf, actb);

  // ---- stage B: cross-attention ----
  k_gemm<__bf16, false><<<gDD, 256, 0, stream>>>(actb, wcab[0], nullptr, qb, ROWS, DMODEL, DMODEL);
  k_gemm<__bf16, false><<<gDD, 256, 0, stream>>>(encb, wcab[1], nullptr, kb, ROWS, DMODEL, DMODEL);
  k_gemm<__bf16, false><<<gDD, 256, 0, stream>>>(encb, wcab[2], nullptr, vb, ROWS, DMODEL, DMODEL);
  k_transpose_v<<<(unsigned)(EA / 256), 256, 0, stream>>>(vb, vtb);
  k_attn<false><<<BATCH * N_HEADS * (S_LEN / 16) / 8, 256, 0, stream>>>(qb, kb, vtb, attnb);
  k_gemm<float, false><<<gDD, 256, 0, stream>>>(attnb, wcab[3], nullptr, tmpf, ROWS, DMODEL, DMODEL);
  k_add_ln<<<ROWS, 256, 0, stream>>>(hf, tmpf, ln2g, ln2b, hf, actb);

  // ---- stage C: feed-forward ----
  k_gemm<__bf16, true ><<<gDH, 256, 0, stream>>>(actb, w1b, ffb1, ff1, ROWS, D_HID, DMODEL);
  k_gemm<float, false><<<gDD, 256, 0, stream>>>(ff1, w2b, ffb2, tmpf, ROWS, DMODEL, D_HID);
  k_add_ln<<<ROWS, 256, 0, stream>>>(hf, tmpf, ln3g, ln3b, (float*)d_out, nullptr);
}
